// LiveNet_81071802679749
// MI455X (gfx1250) — matrix operand order repaired and checked by reading, not compiled, not run on hardware
//
#include <hip/hip_runtime.h>

typedef __fp16   v2hf __attribute__((ext_vector_type(2)));   // return type of cvt_pkrtz
typedef _Float16 v4h  __attribute__((ext_vector_type(4)));
typedef _Float16 v8h  __attribute__((ext_vector_type(8)));
typedef _Float16 v16h __attribute__((ext_vector_type(16)));
typedef float    v8f  __attribute__((ext_vector_type(8)));
typedef unsigned int u32x4 __attribute__((ext_vector_type(4)));
typedef int      i32x4 __attribute__((ext_vector_type(4)));
typedef int      i32x8 __attribute__((ext_vector_type(8)));

#define TILE_M   128      // rows of x per workgroup
#define KC       128      // K-chunk staged in LDS for layer 1
#define NCHUNK   (512 / KC)
#define D_IN     512
#define D_MID    128
#define D_OUT    64
#define SSTRIDE  136      // padded LDS row stride in halves -> 68 dwords, bank-spread

union FragU { v16h v; v8h h[2]; };
union PackU { v4h v; v2hf p[2]; };   // bit-cast __fp16x2 pair -> _Float16x4

// float4 -> 4 x f16 using v_cvt_pkrtz_f16_f32
__device__ __forceinline__ v4h cvt4(const float4 f) {
    PackU u;
    u.p[0] = __builtin_amdgcn_cvt_pkrtz(f.x, f.y);
    u.p[1] = __builtin_amdgcn_cvt_pkrtz(f.z, f.w);
    return u.v;
}

// ---- staging helpers: 256 threads, 128x128 f32 tile, thread (wave,lane) owns rows wave+8j, cols lane*4 ----
__device__ __forceinline__ void fetch_x(const float* __restrict__ g, float4* rx, int wave, int lane) {
    #pragma unroll
    for (int j = 0; j < 16; ++j)
        rx[j] = *reinterpret_cast<const float4*>(g + (size_t)(wave + 8 * j) * D_IN + lane * 4);
}
__device__ __forceinline__ void commit_x(const float4* rx, _Float16* __restrict__ s, int wave, int lane) {
    #pragma unroll
    for (int j = 0; j < 16; ++j)
        *reinterpret_cast<v4h*>(s + (wave + 8 * j) * SSTRIDE + lane * 4) = cvt4(rx[j]);
}
__device__ __forceinline__ void stage_w1(const float* __restrict__ g, _Float16* __restrict__ s,
                                         int wave, int lane) {
    #pragma unroll
    for (int j = 0; j < 16; ++j) {
        const float4 v = *reinterpret_cast<const float4*>(g + (size_t)(wave + 8 * j) * D_IN + lane * 4);
        *reinterpret_cast<v4h*>(s + (wave + 8 * j) * SSTRIDE + lane * 4) = cvt4(v);
    }
}
__device__ __forceinline__ void stage_w2(const float* __restrict__ g, _Float16* __restrict__ s,
                                         int wave, int lane) {
    #pragma unroll
    for (int j = 0; j < 8; ++j) {   // 64 rows x 128 cols
        const float4 v = *reinterpret_cast<const float4*>(g + (size_t)(wave + 8 * j) * D_MID + lane * 4);
        *reinterpret_cast<v4h*>(s + (wave + 8 * j) * SSTRIDE + lane * 4) = cvt4(v);
    }
}

// ---- TDM: DMA one 128(rows) x 128(cols) f16 W1 chunk from global (row stride 512 halves)
//      into LDS with 16B padding after every 256B row (-> SSTRIDE=136 halves) ----
__device__ __forceinline__ void tdm_load_w1_chunk(const _Float16* gsrc, _Float16* ldst) {
    const unsigned long long ga = (unsigned long long)(uintptr_t)gsrc;
    u32x4 g0;
    g0.x = 1u;                                               // count=1 valid descriptor
    g0.y = (unsigned)(uintptr_t)ldst;                        // lds_addr = low 32 bits of flat LDS addr
    g0.z = (unsigned)ga;                                     // global_addr[31:0]
    g0.w = (unsigned)((ga >> 32) & 0x01FFFFFFu) | (2u << 30);// global_addr[56:32], type=2 (image)
    i32x8 g1;
    g1[0] = (1 << 16)            // data_size = 1 -> 2 bytes
          | (1 << 20)            // pad_enable
          | (5 << 22)            // pad_interval: 64 DWORDs (= one 256B tile row)
          | (3 << 25);           // pad_amount: 4 DWORDs (= 16B -> row stride 272B)
    g1[1] = (int)((D_IN & 0xFFFF) << 16);                    // tensor_dim0[15:0] (bits 63:48)
    g1[2] = (int)((D_IN >> 16) | ((D_MID & 0xFFFF) << 16));  // tensor_dim0 hi | tensor_dim1 lo
    g1[3] = (int)((D_MID >> 16) | (KC << 16));               // tensor_dim1 hi | tile_dim0 = 128
    g1[4] = (int)D_MID;                                      // tile_dim1 = 128 rows, tile_dim2 = 0
    g1[5] = (int)D_IN;                                       // tensor_dim0_stride = 512 elements
    g1[6] = 0;                                               // stride0 hi | stride1 lo (unused, 2D)
    g1[7] = 0;                                               // stride1 hi (unused)
    const i32x4 z4 = {0, 0, 0, 0};
#if __clang_major__ >= 23
    const i32x8 z8 = {0, 0, 0, 0, 0, 0, 0, 0};
    __builtin_amdgcn_tensor_load_to_lds(g0, g1, z4, z4, z8, 0);
#else
    __builtin_amdgcn_tensor_load_to_lds(g0, g1, z4, z4, 0);
#endif
}

// ---- one 128-wide K-chunk of layer 1, 2 M-tiles x 4 N-tiles per wave (12 LDS loads / 8 WMMAs) ----
__device__ __forceinline__ void compute_chunk(const _Float16* __restrict__ xs,
                                              const _Float16* __restrict__ w1s,
                                              v8f acc[8], int rowA0, int colB0, int lrow, int half) {
    #pragma unroll
    for (int ks = 0; ks < KC / 32; ++ks) {
        const int kk = ks * 32;
        FragU a[2];
        #pragma unroll
        for (int mi = 0; mi < 2; ++mi) {
            const int r = rowA0 + mi * 16;
            // A 16x32 f16 layout: v0..3 -> K = half*8 + 0..7 ; v4..7 -> K = 16 + half*8 + 0..7
            a[mi].h[0] = *reinterpret_cast<const v8h*>(&xs[r * SSTRIDE + kk + half * 8]);
            a[mi].h[1] = *reinterpret_cast<const v8h*>(&xs[r * SSTRIDE + kk + 16 + half * 8]);
        }
        FragU b[4];
        #pragma unroll
        for (int t = 0; t < 4; ++t) {
            const int n = colB0 + t * 16 + lrow;
            // B 32x16 f16 layout: v0..3 -> K = half*8 + 0..7 ; v4..7 -> K = 16 + half*8 + 0..7
            b[t].h[0] = *reinterpret_cast<const v8h*>(&w1s[n * SSTRIDE + kk + half * 8]);
            b[t].h[1] = *reinterpret_cast<const v8h*>(&w1s[n * SSTRIDE + kk + 16 + half * 8]);
        }
        #pragma unroll
        for (int mi = 0; mi < 2; ++mi)
            #pragma unroll
            for (int t = 0; t < 4; ++t)
                acc[mi * 4 + t] = __builtin_amdgcn_wmma_f32_16x16x32_f16(
                    false, a[mi].v, false, b[t].v, (short)0, acc[mi * 4 + t], false, false);
    }
}

template<bool USE_TDM>
__global__ void __launch_bounds__(256, 1)
livenet_mlp_fused(const float* __restrict__ x,  const float* __restrict__ W1,
                  const float* __restrict__ b1, const float* __restrict__ W2,
                  const float* __restrict__ b2, float* __restrict__ y,
                  const _Float16* __restrict__ w1h) {
    __shared__ __align__(16) _Float16 xs [2][TILE_M * SSTRIDE];  // double-buffered x chunk, f16
    __shared__ __align__(16) _Float16 w1s[2][D_MID  * SSTRIDE];  // double-buffered W1 chunk, f16
    __shared__ __align__(16) _Float16 hs [TILE_M * SSTRIDE];     // hidden activations, f16
    __shared__ __align__(16) _Float16 w2s[D_OUT  * SSTRIDE];     // W2 full, f16
    __shared__ float b1s[D_MID];
    __shared__ float b2s[D_OUT];

    const int tid  = threadIdx.x;
    const int lane = tid & 31;
    const int wave = tid >> 5;
    const int lrow = lane & 15;
    const int half = lane >> 4;
    const int rowA0 = (wave & 3) * 32 + lrow;     // layer-1: wave owns M-tiles 2*(wave&3), +1
    const int colB0 = (wave >> 2) * 64;           // layer-1: wave owns N-tiles 4*(wave>>2)..+3
    const size_t rowbase = (size_t)blockIdx.x * TILE_M;
    const float* xg = x + rowbase * D_IN;

    // Prologue: stage chunk 0 + weights/biases.
    float4 rx[16];
    if (USE_TDM) {
        if (wave == 0) tdm_load_w1_chunk(w1h, &w1s[0][0]);     // DMA, TENSORcnt++
    } else {
        stage_w1(W1, &w1s[0][0], wave, lane);
    }
    fetch_x(xg, rx, wave, lane);
    stage_w2(W2, w2s, wave, lane);
    if (tid < D_MID) b1s[tid] = b1[tid];
    if (tid < D_OUT) b2s[tid] = b2[tid];
    commit_x(rx, &xs[0][0], wave, lane);
    if (USE_TDM && wave == 0) __builtin_amdgcn_s_wait_tensorcnt(0);
    __syncthreads();

    v8f acc[8];
    #pragma unroll
    for (int t = 0; t < 8; ++t) acc[t] = v8f{0.f,0.f,0.f,0.f,0.f,0.f,0.f,0.f};

    // -------- Layer 1: double-buffered; x fetch + W1 TDM for c+1 run under compute of c --------
    for (int c = 0; c < NCHUNK; ++c) {
        const int cur = c & 1, nxt = cur ^ 1;
        if (c < NCHUNK - 1) {
            if (USE_TDM) {
                if (wave == 0) tdm_load_w1_chunk(w1h + (c + 1) * KC, &w1s[nxt][0]);
            }
            fetch_x(xg + (c + 1) * KC, rx, wave, lane);        // global loads in flight
        }
        compute_chunk(&xs[cur][0], &w1s[cur][0], acc, rowA0, colB0, lrow, half);
        if (c < NCHUNK - 1) {
            if (!USE_TDM) stage_w1(W1 + (c + 1) * KC, &w1s[nxt][0], wave, lane);
            commit_x(rx, &xs[nxt][0], wave, lane);             // wait loads, cvt, ds_store
            if (USE_TDM && wave == 0) __builtin_amdgcn_s_wait_tensorcnt(0);
        }
        __syncthreads();   // next buffers staged (incl. TDM complete); current buffers free
    }

    // Bias + ReLU -> hs (f16, row-major). C/D layout: VGPR r holds M = 8*half + r, N = 16t + lrow.
    #pragma unroll
    for (int mi = 0; mi < 2; ++mi) {
        #pragma unroll
        for (int t = 0; t < 4; ++t) {
            const int n = colB0 + t * 16 + lrow;
            const float bias = b1s[n];
            #pragma unroll
            for (int r = 0; r < 8; ++r) {
                float hv = acc[mi * 4 + t][r] + bias;
                hv = hv > 0.f ? hv : 0.f;
                hs[((wave & 3) * 32 + mi * 16 + half * 8 + r) * SSTRIDE + n] = (_Float16)hv;
            }
        }
    }
    __syncthreads();   // layer 2 reads rows written by other waves

    // -------- Layer 2: y = relu(h @ W2^T + b2), K = 128; wave owns rows 16*wave..+15 --------
    const int mrow = wave * 16 + lrow;
    v8f acc2[4];
    #pragma unroll
    for (int t = 0; t < 4; ++t) acc2[t] = v8f{0.f,0.f,0.f,0.f,0.f,0.f,0.f,0.f};

    #pragma unroll
    for (int ks = 0; ks < D_MID / 32; ++ks) {
        const int kk = ks * 32;
        FragU a;
        a.h[0] = *reinterpret_cast<const v8h*>(&hs[mrow * SSTRIDE + kk + half * 8]);
        a.h[1] = *reinterpret_cast<const v8h*>(&hs[mrow * SSTRIDE + kk + 16 + half * 8]);
        FragU b[4];
        #pragma unroll
        for (int t = 0; t < 4; ++t) {
            const int n = t * 16 + lrow;
            b[t].h[0] = *reinterpret_cast<const v8h*>(&w2s[n * SSTRIDE + kk + half * 8]);
            b[t].h[1] = *reinterpret_cast<const v8h*>(&w2s[n * SSTRIDE + kk + 16 + half * 8]);
        }
        #pragma unroll
        for (int t = 0; t < 4; ++t)
            acc2[t] = __builtin_amdgcn_wmma_f32_16x16x32_f16(
                false, a.v, false, b[t].v, (short)0, acc2[t], false, false);
    }

    // Bias + ReLU -> global y [B, 64] f32
    #pragma unroll
    for (int t = 0; t < 4; ++t) {
        const int n = t * 16 + lrow;
        const float bias = b2s[n];
        #pragma unroll
        for (int r = 0; r < 8; ++r) {
            float yv = acc2[t][r] + bias;
            yv = yv > 0.f ? yv : 0.f;
            y[(rowbase + wave * 16 + half * 8 + r) * D_OUT + n] = yv;
        }
    }
}

// ---- pre-pass: convert W1 (128x512 f32) to f16 once into workspace ----
__global__ void __launch_bounds__(256)
cvt_w1_kernel(const float* __restrict__ W1, _Float16* __restrict__ w1h) {
    const int i = (blockIdx.x * 256 + threadIdx.x) * 4;   // 65536 elements total
    const float4 v = *reinterpret_cast<const float4*>(W1 + i);
    *reinterpret_cast<v4h*>(w1h + i) = cvt4(v);
}

extern "C" void kernel_launch(void* const* d_in, const int* in_sizes, int n_in,
                              void* d_out, int out_size, void* d_ws, size_t ws_size,
                              hipStream_t stream) {
    const float* x  = (const float*)d_in[0];
    const float* W1 = (const float*)d_in[1];
    const float* b1 = (const float*)d_in[2];
    const float* W2 = (const float*)d_in[3];
    const float* b2 = (const float*)d_in[4];
    float* y = (float*)d_out;

    const int B = 65536;
    dim3 grid(B / TILE_M);   // 512 workgroups
    dim3 block(256);         // 8 wave32

    const size_t w1h_bytes = (size_t)D_MID * D_IN * sizeof(_Float16);  // 128 KB
    if (ws_size >= w1h_bytes) {
        _Float16* w1h = (_Float16*)d_ws;
        cvt_w1_kernel<<<dim3(D_MID * D_IN / (256 * 4)), block, 0, stream>>>(W1, w1h);
        livenet_mlp_fused<true><<<grid, block, 0, stream>>>(x, W1, b1, W2, b2, y, w1h);
    } else {
        livenet_mlp_fused<false><<<grid, block, 0, stream>>>(x, W1, b1, W2, b2, y, nullptr);
    }
}
